// CapsuleLayer_6287832122042
// MI455X (gfx1250) — compile-verified, matrix operand order checked by reading, not run on hardware
//
#include <hip/hip_runtime.h>
#include <hip/hip_bf16.h>

typedef __attribute__((ext_vector_type(16))) _Float16 v16h;
typedef __attribute__((ext_vector_type(8)))  float    v8f;

constexpr int B_ = 256, I_ = 1152, D_ = 8, C_ = 10, O_ = 16;
constexpr int N_ = C_ * O_;          // 160
constexpr int CHUNK  = 128;          // i-rows staged per chunk in routing
constexpr int NCHUNK = I_ / CHUNK;   // 9
constexpr int NROUT  = 3;
constexpr int HSROW  = N_ + 8;       // 168 f16 = 336B padded row (2-way vs 8-way bank conflicts)

#if __has_builtin(__builtin_amdgcn_global_load_async_to_lds_b128) && \
    __has_builtin(__builtin_amdgcn_s_wait_asynccnt)
#define USE_ASYNC_LDS 1
#else
#define USE_ASYNC_LDS 0
#endif

#if USE_ASYNC_LDS
// The builtin wants int4* in AS(1)/AS(3) (per hipcc diagnostic).
typedef int v4i __attribute__((ext_vector_type(4)));
typedef __attribute__((address_space(1))) v4i* g1v4i;
typedef __attribute__((address_space(3))) v4i* l3v4i;
__device__ __forceinline__ g1v4i to_global(const void* p) {
  return (g1v4i)(uintptr_t)p;                 // generic->AS(1): same 64-bit value
}
__device__ __forceinline__ l3v4i to_lds(void* p) {
  return (l3v4i)(uintptr_t)p;                 // generic->AS(3): low 32 bits = LDS offset
}
#endif

// -----------------------------------------------------------------------------
// Kernel 1: hat[b,i,n] (f16, n = c*16+o) = sum_d x[b,i,d] * W[i,c,d,o]
// Per i: [256x8]x[8x160] GEMM -> WMMA f32_16x16x32_f16, K padded 8->32.
// grid = (1152 i, 4 batch-groups of 64), block = 128 (4 waves, 1 M-tile each).
// -----------------------------------------------------------------------------
__global__ __launch_bounds__(128) void caps_hat_wmma(const float* __restrict__ X,
                                                     const float* __restrict__ Wt,
                                                     _Float16* __restrict__ HAT) {
  const int i    = blockIdx.x;
  const int bb   = blockIdx.y * 64;          // first batch row of this block
  const int tid  = threadIdx.x;
  const int w    = tid >> 5;                 // wave 0..3 -> M-tile
  const int lane = tid & 31;
  const int l16  = lane & 15;
  const bool lo  = lane < 16;

  __shared__ _Float16 hbuf[64 * N_];         // 20 KB staging for coalesced store

  // A fragment (16-bit A 16x32): lanes 0-15 = rows M, elements 0..7 = K(d)=0..7,
  // lanes 16-31 hold K=8..15 -> zero padding. Elements 8..15 (K>=16) zero.
  v16h a;
  #pragma unroll
  for (int j = 0; j < 16; ++j) a[j] = (_Float16)0.0f;
  if (lo) {
    const float* xp = X + ((size_t)(bb + w * 16 + l16) * I_ + i) * D_;
    #pragma unroll
    for (int d = 0; d < D_; ++d) a[d] = (_Float16)xp[d];
  }

  const float* wp_i = Wt + (size_t)i * (C_ * D_ * O_);

  #pragma unroll
  for (int c = 0; c < C_; ++c) {
    // B fragment (16-bit B 32x16): lanes 0-15 = column o, elements 0..7 = K=0..7
    // (d=0..7 live; elements 8..15 = K=16..23, zero); lanes 16-31 hold K=8..15,24..31 -> zero.
    v16h bf;
    #pragma unroll
    for (int j = 0; j < 16; ++j) bf[j] = (_Float16)0.0f;
    if (lo) {
      const float* wp = wp_i + c * (D_ * O_) + l16;     // W[i,c,d,o=l16]
      #pragma unroll
      for (int d = 0; d < D_; ++d) bf[d] = (_Float16)wp[d * O_];
    }
    v8f acc;
    #pragma unroll
    for (int r = 0; r < 8; ++r) acc[r] = 0.0f;
    acc = __builtin_amdgcn_wmma_f32_16x16x32_f16(
        /*neg_a=*/false, a, /*neg_b=*/false, bf,
        /*c_mod=*/(short)0, acc, /*reuse_a=*/false, /*reuse_b=*/false);

    // C/D layout: VGPR r, lanes 0-15: (M=r, N=lane); lanes 16-31: (M=r+8, N=lane-16)
    const int Mh = (lane >> 4) * 8;
    const int Nn = lane & 15;
    #pragma unroll
    for (int r = 0; r < 8; ++r)
      hbuf[(w * 16 + Mh + r) * N_ + c * O_ + Nn] = (_Float16)acc[r];
  }
  __syncthreads();

  // Coalesced f16 store: 64 rows x 320B -> 1280 chunks of 16B, 10 per thread.
  #pragma unroll
  for (int j = 0; j < 10; ++j) {
    const int k   = tid + 128 * j;
    const int bl  = k / 20;
    const int off = (k % 20) * 8;            // f16 elements
    const uint4 v = *(const uint4*)&hbuf[bl * N_ + off];
    *(uint4*)&HAT[((size_t)(bb + bl) * I_ + i) * N_ + off] = v;
  }
}

// -----------------------------------------------------------------------------
// Kernel 2: all 3 routing iterations for one sample b per block.
// bias is never stored: bias_r[i,c] = sum_{r'<r} dot_o(hat[i,c,:], out_{r'}[c,:])
// recomputed from the tiny per-iteration outputs kept in LDS.
// Chunk staging uses GLOBAL_LOAD_ASYNC_TO_LDS_B128 (ASYNCcnt) when available.
// block = 256 threads; thread n<160 owns s[c=n/16, o=n%16].
// -----------------------------------------------------------------------------
__global__ __launch_bounds__(256) void caps_route(const _Float16* __restrict__ HAT,
                                                  float* __restrict__ OUT) {
  const int b   = blockIdx.x;
  const int tid = threadIdx.x;
  const int n   = tid;
  const int c16 = (n >> 4) << 4;

  __shared__ _Float16 hs[CHUNK * HSROW];     // 43 KB padded hat chunk
  __shared__ float    cw[CHUNK * C_];        // softmax weights per staged i
  __shared__ float    outp[NROUT * N_];      // per-iteration outputs
  __shared__ float    sb[N_];

  float out_n = 0.0f;

  for (int r = 0; r < NROUT; ++r) {
    float s_acc = 0.0f;
    for (int ch = 0; ch < NCHUNK; ++ch) {
      __syncthreads();                       // hs/cw reuse + outp visibility
      // ---- stage 128 i-rows (40960 B, 2560x16B; 10 chunks/thread) ----
      const uint4* src = (const uint4*)(HAT + ((size_t)b * I_ + (size_t)ch * CHUNK) * N_);
#if USE_ASYNC_LDS
      #pragma unroll
      for (int j = 0; j < 10; ++j) {
        const int k   = tid + 256 * j;
        const int bl  = k / 20;              // row within chunk
        const int off = (k % 20) * 8;        // f16 elements within row
        __builtin_amdgcn_global_load_async_to_lds_b128(
            to_global(src + k), to_lds(&hs[bl * HSROW + off]), 0, 0);
      }
      __builtin_amdgcn_s_wait_asynccnt(0);
#else
      #pragma unroll
      for (int j = 0; j < 10; ++j) {
        const int k   = tid + 256 * j;
        const int bl  = k / 20;
        const int off = (k % 20) * 8;
        *(uint4*)&hs[bl * HSROW + off] = src[k];
      }
#endif
      __syncthreads();
      // warm L2 for the next chunk while we compute on this one
      if (ch + 1 < NCHUNK)
        __builtin_prefetch((const void*)(src + 2560 + (size_t)tid * 10), 0, 0);
      // ---- phase A: per-i routing logits -> softmax over c ----
      if (tid < CHUNK) {
        float bias[C_];
        #pragma unroll
        for (int c = 0; c < C_; ++c) bias[c] = 0.0f;
        for (int rp = 0; rp < r; ++rp) {
          #pragma unroll
          for (int c = 0; c < C_; ++c) {
            const _Float16* hp = &hs[tid * HSROW + c * O_];
            float acc = 0.0f;
            #pragma unroll
            for (int o = 0; o < O_; ++o)
              acc += (float)hp[o] * outp[rp * N_ + c * O_ + o];
            bias[c] += acc;
          }
        }
        float m = bias[0];
        #pragma unroll
        for (int c = 1; c < C_; ++c) m = fmaxf(m, bias[c]);
        float e[C_];
        float sum = 0.0f;
        #pragma unroll
        for (int c = 0; c < C_; ++c) { e[c] = __expf(bias[c] - m); sum += e[c]; }
        const float inv = 1.0f / sum;
        #pragma unroll
        for (int c = 0; c < C_; ++c) cw[tid * C_ + c] = e[c] * inv;
      }
      __syncthreads();
      // ---- phase B: s[n] += cw[i,c] * hat[i,n] over staged i ----
      if (n < N_) {
        const int c = n >> 4;
        for (int il = 0; il < CHUNK; ++il)
          s_acc += cw[il * C_ + c] * (float)hs[il * HSROW + n];
      }
    }
    // ---- squash: v = (|s|^2/(1+|s|^2)) * s / sqrt(|s|^2+eps) over o ----
    __syncthreads();
    if (n < N_) sb[n] = s_acc;
    __syncthreads();
    if (n < N_) {
      float n2 = 0.0f;
      #pragma unroll
      for (int o = 0; o < O_; ++o) { const float v = sb[c16 + o]; n2 += v * v; }
      const float scale = n2 / ((1.0f + n2) * sqrtf(n2 + 1e-7f));
      out_n = s_acc * scale;
      outp[r * N_ + n] = out_n;              // consumed by next pass's phase A
    }
  }
  if (n < N_) OUT[(size_t)b * N_ + n] = out_n;
}

// -----------------------------------------------------------------------------
// Launch: hat lives in d_ws as f16: 256*1152*160*2 = 94,371,840 bytes
// (fits the 192 MB L2 -> routing passes stream from L2, not HBM).
// -----------------------------------------------------------------------------
extern "C" void kernel_launch(void* const* d_in, const int* in_sizes, int n_in,
                              void* d_out, int out_size, void* d_ws, size_t ws_size,
                              hipStream_t stream) {
  (void)in_sizes; (void)n_in; (void)out_size; (void)ws_size;
  const float* x = (const float*)d_in[0];    // [B, I, D] f32
  const float* W = (const float*)d_in[1];    // [I, C, D, O] f32
  float* out     = (float*)d_out;            // [B, C, O] f32
  _Float16* hat  = (_Float16*)d_ws;          // [B, I, C*O] f16

  caps_hat_wmma<<<dim3(I_, 4, 1), 128, 0, stream>>>(x, W, hat);
  caps_route<<<dim3(B_, 1, 1), 256, 0, stream>>>(hat, out);
}
